// StackedLSTM_81862076662481
// MI455X (gfx1250) — matrix operand order repaired and checked by reading, not compiled, not run on hardware
//
#include <hip/hip_runtime.h>
#include <hip/hip_bf16.h>

// ---------------------------------------------------------------------------
// Stacked 2-layer LSTM on gfx1250 (MI455X), wave32 + WMMA bf16.
//   S=512, B=64, N=H=512.  g = [x|h](64x1024) @ Wcat(1024x2048) + 2b
//   A is staged in LDS in WMMA A-fragment order (32B/lane/K-step) so the
//   inner loop is: 2x ds_load_b128 (A) + 8x global_load_b128 (B, 4 gates)
//   + 4x v_wmma_f32_16x16x32_bf16, with strided pointer addressing.
// ---------------------------------------------------------------------------

#define SEQ   512
#define BATCH 64
#define HID   512   // == N
#define KTOT  1024  // N + H
#define GDIM  2048  // 4*H
#define KSTEPS (KTOT / 32)          // 32
#define WSTRIDE (128 * 512)         // packed elems per K-step (128 tiles x 512)

typedef __attribute__((ext_vector_type(16))) __bf16          v16bf;
typedef __attribute__((ext_vector_type(16))) unsigned short  v16u;
typedef __attribute__((ext_vector_type(8)))  float           v8f;

static __device__ __forceinline__ unsigned short f2bf(float f) {
  unsigned u = __builtin_bit_cast(unsigned, f);
  unsigned r = u + 0x7FFFu + ((u >> 16) & 1u);   // round-to-nearest-even
  return (unsigned short)(r >> 16);
}
static __device__ __forceinline__ float sigm(float v) {
  return 1.0f / (1.0f + __expf(-v));
}
static __device__ __forceinline__ float tanh_fast(float v) {
  float e = __expf(2.0f * v);
  return 1.0f - 2.0f / (e + 1.0f);     // robust at +/-inf
}

// ---------------------------------------------------------------------------
// Pack Wcat = [Wx ; Wh] (f32, row-major KTOT x GDIM) into bf16 WMMA B-fragment
// tile order: tile(kt,nt) (32x16), within tile lane-major, 16 contiguous
// ushorts per lane:  k = kt*32 + ((e>>3)<<4) + ((lane>>4)<<3) + (e&7),  n = nt*16 + (lane&15).
// Also bias2 = 2*b (reference adds bias in both addmm calls).
// ---------------------------------------------------------------------------
__global__ __launch_bounds__(256)
void pack_weights_kernel(const float* __restrict__ Wx, const float* __restrict__ Wh,
                         const float* __restrict__ b,  unsigned short* __restrict__ Wp,
                         float* __restrict__ bias2) {
  size_t p = (size_t)blockIdx.x * 256 + threadIdx.x;      // 0 .. KTOT*GDIM-1
  if (p < GDIM) bias2[p] = 2.0f * b[p];
  if (p >= (size_t)KTOT * GDIM) return;

  unsigned tile   = (unsigned)(p >> 9);      // /512 elems per 32x16 tile
  unsigned within = (unsigned)(p & 511u);
  unsigned lane   = within >> 4;
  unsigned e      = within & 15u;
  unsigned kt     = tile >> 7;               // 0..31   (KTOT/32)
  unsigned nt     = tile & 127u;             // 0..127  (GDIM/16)
  unsigned k      = kt * 32 + ((e >> 3) << 4) + ((lane >> 4) << 3) + (e & 7u);
  unsigned n      = nt * 16 + (lane & 15u);

  float v = (k < HID) ? Wx[(size_t)k * GDIM + n]
                      : Wh[(size_t)(k - HID) * GDIM + n];
  Wp[p] = f2bf(v);
}

// ---------------------------------------------------------------------------
// One LSTM cell step.  grid = 16 blocks x 256 threads (8 waves).
//   block: mt = blockIdx>>2 (16 batch rows), jgrp = blockIdx&3
//   wave:  jt = jgrp*8 + waveId  -> gate-column block j = jt*16 .. +15
//   acc0..acc3 = f, i1, i2, o tiles for the SAME (row, j) block.
// ---------------------------------------------------------------------------
__global__ __launch_bounds__(256)
void lstm_cell_kernel(const float* __restrict__ x,     // [B, HID]
                      const float* __restrict__ hin,   // [B, HID]
                      float* __restrict__ cio,         // [B, HID] in/out
                      float* __restrict__ hout,        // [B, HID]
                      const unsigned short* __restrict__ Wp,  // packed bf16
                      const float* __restrict__ bias2) {      // [GDIM]
  // A staged in WMMA fragment order: Af[(ks*32 + lane)*16 + e], 32 KiB.
  __shared__ unsigned short Af[KSTEPS * 32 * 16];

  const int tid  = threadIdx.x;
  const int lane = tid & 31;
  const int wv   = tid >> 5;                 // 0..7
  const int mt   = blockIdx.x >> 2;          // 0..3
  const int jt   = (blockIdx.x & 3) * 8 + wv;// 0..31
  const int rowbase = mt * 16;

  // Stage A = [x | h] rows rowbase..+15 as bf16, scattered into fragment
  // order.  Fragment map: k = ks*32 + ((e>>3)<<4) + ((lane>>4)<<3) + (e&7),
  // row = lane&15.  Inverse: e = ((k>>4)&1)<<3 | (k&7), lane = ((k>>3)&1)<<4 | row.
  for (int idx = tid; idx < 16 * KTOT; idx += 256) {
    int m = idx >> 10, k = idx & (KTOT - 1);
    float v = (k < HID) ? x  [(size_t)(rowbase + m) * HID + k]
                        : hin[(size_t)(rowbase + m) * HID + (k - HID)];
    int ks = k >> 5;
    int e  = (((k >> 4) & 1) << 3) | (k & 7);
    int ln = (((k >> 3) & 1) << 4) | m;
    Af[(ks * 32 + ln) * 16 + e] = f2bf(v);
  }
  __syncthreads();

  v8f acc0 = {}, acc1 = {}, acc2 = {}, acc3 = {};

  // Loop-carried strided pointers (stride = WSTRIDE elems per K-step).
  const unsigned short* pA = &Af[lane * 16];
  const unsigned short* p0 = Wp + ((size_t)(  0 + jt) * 32 + lane) * 16;
  const unsigned short* p1 = Wp + ((size_t)( 32 + jt) * 32 + lane) * 16;
  const unsigned short* p2 = Wp + ((size_t)( 64 + jt) * 32 + lane) * 16;
  const unsigned short* p3 = Wp + ((size_t)( 96 + jt) * 32 + lane) * 16;

  for (int ks = 0; ks < KSTEPS; ++ks) {
    v16bf a = __builtin_bit_cast(v16bf, *(const v16u*)pA);   // 2x ds_load_b128
    pA += 32 * 16;

    if (ks + 1 < KSTEPS) {   // gfx1250 global_prefetch for next K-step tile
      __builtin_prefetch(p0 + WSTRIDE, 0, 1);
    }

    v16u b0v = *(const v16u*)p0;  p0 += WSTRIDE;
    v16u b1v = *(const v16u*)p1;  p1 += WSTRIDE;
    v16u b2v = *(const v16u*)p2;  p2 += WSTRIDE;
    v16u b3v = *(const v16u*)p3;  p3 += WSTRIDE;

    acc0 = __builtin_amdgcn_wmma_f32_16x16x32_bf16(false, a, false,
             __builtin_bit_cast(v16bf, b0v), (short)0, acc0, false, false);
    acc1 = __builtin_amdgcn_wmma_f32_16x16x32_bf16(false, a, false,
             __builtin_bit_cast(v16bf, b1v), (short)0, acc1, false, false);
    acc2 = __builtin_amdgcn_wmma_f32_16x16x32_bf16(false, a, false,
             __builtin_bit_cast(v16bf, b2v), (short)0, acc2, false, false);
    acc3 = __builtin_amdgcn_wmma_f32_16x16x32_bf16(false, a, false,
             __builtin_bit_cast(v16bf, b3v), (short)0, acc3, false, false);
  }

  // Epilogue: C/D layout -> element v: m = v + 8*(lane>=16), n = lane&15.
  const int j    = jt * 16 + (lane & 15);
  const int mofs = (lane >> 4) << 3;
#pragma unroll
  for (int v = 0; v < 8; ++v) {
    int m = rowbase + v + mofs;
    size_t idx = (size_t)m * HID + j;
    float f  = acc0[v] + bias2[j];
    float i1 = acc1[v] + bias2[HID + j];
    float i2 = acc2[v] + bias2[2 * HID + j];
    float o  = acc3[v] + bias2[3 * HID + j];
    float cn = sigm(f) * cio[idx] + sigm(i1) * tanh_fast(i2);
    cio[idx]  = cn;
    hout[idx] = sigm(o) * tanh_fast(cn);
  }
}

// ---------------------------------------------------------------------------
extern "C" void kernel_launch(void* const* d_in, const int* in_sizes, int n_in,
                              void* d_out, int out_size, void* d_ws, size_t ws_size,
                              hipStream_t stream) {
  const float* inputs = (const float*)d_in[0];   // [S, B, N]
  const float* Wx0    = (const float*)d_in[1];   // [N, 4H]
  const float* Wh0    = (const float*)d_in[2];   // [H, 4H]
  const float* b0     = (const float*)d_in[3];   // [4H]
  const float* Wx1    = (const float*)d_in[4];
  const float* Wh1    = (const float*)d_in[5];
  const float* b1     = (const float*)d_in[6];
  float* out          = (float*)d_out;           // [S, B, H]

  const size_t WP_ELEMS = (size_t)KTOT * GDIM;       // 2M bf16
  const size_t BH       = (size_t)BATCH * HID;       // 32K f32

  char* ws = (char*)d_ws;
  unsigned short* Wp0 = (unsigned short*)ws;  ws += WP_ELEMS * 2;
  unsigned short* Wp1 = (unsigned short*)ws;  ws += WP_ELEMS * 2;
  float* bias20 = (float*)ws;  ws += GDIM * 4;
  float* bias21 = (float*)ws;  ws += GDIM * 4;
  float* h0a    = (float*)ws;  ws += BH * 4;
  float* h0b    = (float*)ws;  ws += BH * 4;
  float* c0     = (float*)ws;  ws += BH * 4;
  float* c1     = (float*)ws;  ws += BH * 4;
  float* zbuf   = (float*)ws;  ws += BH * 4;

  // Re-zero recurrent state every call (deterministic under graph replay).
  hipMemsetAsync(h0a,  0, BH * 4, stream);
  hipMemsetAsync(c0,   0, BH * 4, stream);
  hipMemsetAsync(c1,   0, BH * 4, stream);
  hipMemsetAsync(zbuf, 0, BH * 4, stream);

  const int packBlocks = (int)(WP_ELEMS / 256);
  pack_weights_kernel<<<packBlocks, 256, 0, stream>>>(Wx0, Wh0, b0, Wp0, bias20);
  pack_weights_kernel<<<packBlocks, 256, 0, stream>>>(Wx1, Wh1, b1, Wp1, bias21);

  float* h0buf[2] = { h0a, h0b };
  for (int t = 0; t < SEQ; ++t) {
    const float* xt    = inputs + (size_t)t * BH;
    const float* h0in  = h0buf[t & 1];
    float*       h0out = h0buf[(t + 1) & 1];
    lstm_cell_kernel<<<16, 256, 0, stream>>>(xt, h0in, c0, h0out, Wp0, bias20);

    const float* h1in  = t ? out + (size_t)(t - 1) * BH : zbuf;
    float*       h1out = out + (size_t)t * BH;            // h1 state lives in d_out
    lstm_cell_kernel<<<16, 256, 0, stream>>>(h0out, h1in, c1, h1out, Wp1, bias21);
  }
}
